// HBClassicNet_18098992185376
// MI455X (gfx1250) — matrix operand order repaired and checked by reading, not compiled, not run on hardware
//
#include <hip/hip_runtime.h>
#include <hip/hip_bf16.h>

typedef __attribute__((ext_vector_type(16))) __bf16 bhalf16;
typedef __attribute__((ext_vector_type(8)))  __bf16 bhalf8;
typedef __attribute__((ext_vector_type(8)))  float  float8;

#define BATCH   32768
#define IN_DIM  784
#define HID     256
#define OUT_DIM 1000

#define BM 128
#define BN 128
#define BK 32
#define LDT 40   // LDS row stride in bf16 elements (padded from 32)

// ---- helpers -------------------------------------------------------------

static __device__ __forceinline__ __bf16 f2bf(float f) {
    unsigned u = __builtin_bit_cast(unsigned, f);
    unsigned r = (u + 0x7FFFu + ((u >> 16) & 1u)) >> 16;
    unsigned short s = (unsigned short)r;
    return __builtin_bit_cast(__bf16, s);
}

static __device__ __forceinline__ bhalf8 pack8(float4 a, float4 b) {
    bhalf8 r;
    r[0] = f2bf(a.x); r[1] = f2bf(a.y); r[2] = f2bf(a.z); r[3] = f2bf(a.w);
    r[4] = f2bf(b.x); r[5] = f2bf(b.y); r[6] = f2bf(b.z); r[7] = f2bf(b.w);
    return r;
}

static __device__ __forceinline__ bhalf16 cat16(bhalf8 lo, bhalf8 hi) {
    bhalf16 r;
#pragma unroll
    for (int i = 0; i < 8; ++i) { r[i] = lo[i]; r[i + 8] = hi[i]; }
    return r;
}

// A fragment (16x32 bf16): lane L -> row m = L&15; hs = L>>4 selects K-half.
// v0..v3 hold K = hs*8 .. hs*8+7 ; v4..v7 hold K = 16+hs*8 .. 16+hs*8+7
static __device__ __forceinline__ bhalf16 load_a_frag(const __bf16* rowp, int hs) {
    bhalf8 lo = *reinterpret_cast<const bhalf8*>(rowp + hs * 8);
    bhalf8 hi = *reinterpret_cast<const bhalf8*>(rowp + 16 + hs * 8);
    return cat16(lo, hi);
}

// B fragment (32x16 bf16), with B[k][n] = W[n][k] stored row-major [n][k]:
// lane L -> n = L&15, hs = L>>4; v0..v3 hold K = hs*8 .. hs*8+7 ; v4..v7 hold K = 16+hs*8 .. 16+hs*8+7
static __device__ __forceinline__ bhalf16 load_b_frag(const __bf16* rowp, int hs) {
    bhalf8 lo = *reinterpret_cast<const bhalf8*>(rowp + hs * 8);
    bhalf8 hi = *reinterpret_cast<const bhalf8*>(rowp + 16 + hs * 8);
    return cat16(lo, hi);
}

// plain f32 -> bf16 staging load: 16 consecutive floats, zero-filled when OOB
static __device__ __forceinline__ void stage16(const float* __restrict__ base,
                                               int row, int gk, int kmax, int stride,
                                               bool rowok, bhalf8& r0, bhalf8& r1) {
    if (rowok && gk < kmax) {
        const float4* p = reinterpret_cast<const float4*>(base + (size_t)row * stride + gk);
        float4 f0 = p[0], f1 = p[1], f2 = p[2], f3 = p[3];
        r0 = pack8(f0, f1);
        r1 = pack8(f2, f3);
    } else {
        r0 = bhalf8{};
        r1 = bhalf8{};
    }
}

// ---- GEMM1: H = X @ W1^T + b1 -------------------------------------------

__global__ __launch_bounds__(256) void gemm1_kernel(const float* __restrict__ X,
                                                    const float* __restrict__ W1,
                                                    const float* __restrict__ B1,
                                                    float* __restrict__ H) {
    // ping-pong LDS: single barrier per k-step
    __shared__ __align__(16) __bf16 As[2][BM * LDT];
    __shared__ __align__(16) __bf16 Bs[2][BN * LDT];

    const int tid  = threadIdx.x;
    const int m0   = blockIdx.x * BM;
    const int n0   = blockIdx.y * BN;
    const int wave = tid >> 5, lane = tid & 31;
    const int wm   = wave >> 1, wn = wave & 1;   // 4 x 2 wave grid, wave tile 32x64
    const int lm   = lane & 15, hs = lane >> 4;

    const int arow = tid >> 1, acb = (tid & 1) * 16;   // 16 floats / thread for A and B
    const int sofs = arow * LDT + acb;

    float8 acc[2][4] = {};
    bhalf8 ra0, ra1, rb0, rb1;

    // pipeline prologue: tile 0 -> regs -> buf0
    stage16(X,  m0 + arow, acb, IN_DIM, IN_DIM, true, ra0, ra1);
    stage16(W1, n0 + arow, acb, IN_DIM, IN_DIM, true, rb0, rb1);
    *reinterpret_cast<bhalf8*>(&As[0][sofs])     = ra0;
    *reinterpret_cast<bhalf8*>(&As[0][sofs + 8]) = ra1;
    *reinterpret_cast<bhalf8*>(&Bs[0][sofs])     = rb0;
    *reinterpret_cast<bhalf8*>(&Bs[0][sofs + 8]) = rb1;
    __syncthreads();

    int cur = 0;
    for (int k0 = 0; k0 < IN_DIM; k0 += BK, cur ^= 1) {
        const int kn = k0 + BK;
        const bool has_next = kn < IN_DIM;

        // issue next tile's global loads before the WMMA burst (latency overlap)
        if (has_next) {
            stage16(X,  m0 + arow, kn + acb, IN_DIM, IN_DIM, true, ra0, ra1);
            stage16(W1, n0 + arow, kn + acb, IN_DIM, IN_DIM, true, rb0, rb1);
        }
        // prefetch tile k+2 toward caches
        const int kp = k0 + 2 * BK;
        if (kp + acb < IN_DIM) {
            __builtin_prefetch(X  + (size_t)(m0 + arow) * IN_DIM + kp + acb, 0, 1);
            __builtin_prefetch(W1 + (size_t)(n0 + arow) * IN_DIM + kp + acb, 0, 1);
        }

        bhalf16 af[2], bf_[4];
#pragma unroll
        for (int i = 0; i < 2; ++i)
            af[i] = load_a_frag(&As[cur][(wm * 32 + i * 16 + lm) * LDT], hs);
#pragma unroll
        for (int j = 0; j < 4; ++j)
            bf_[j] = load_b_frag(&Bs[cur][(wn * 64 + j * 16 + lm) * LDT], hs);

#pragma unroll
        for (int i = 0; i < 2; ++i)
#pragma unroll
            for (int j = 0; j < 4; ++j)
                acc[i][j] = __builtin_amdgcn_wmma_f32_16x16x32_bf16(
                    false, af[i], false, bf_[j], (short)0, acc[i][j], false, false);

        if (has_next) {
            const int nxt = cur ^ 1;
            *reinterpret_cast<bhalf8*>(&As[nxt][sofs])     = ra0;
            *reinterpret_cast<bhalf8*>(&As[nxt][sofs + 8]) = ra1;
            *reinterpret_cast<bhalf8*>(&Bs[nxt][sofs])     = rb0;
            *reinterpret_cast<bhalf8*>(&Bs[nxt][sofs + 8]) = rb1;
            __syncthreads();
        }
    }

#pragma unroll
    for (int i = 0; i < 2; ++i)
#pragma unroll
        for (int j = 0; j < 4; ++j) {
            const int rb = m0 + wm * 32 + i * 16 + hs * 8;
            const int cb = n0 + wn * 64 + j * 16 + lm;
            const float bv = B1[cb];
#pragma unroll
            for (int r = 0; r < 8; ++r)
                H[(size_t)(rb + r) * HID + cb] = acc[i][j][r] + bv;
        }
}

// ---- column stats (two-pass, deterministic) ------------------------------

__global__ __launch_bounds__(256) void colstats_kernel(const float* __restrict__ M,
                                                       float* __restrict__ psum,
                                                       float* __restrict__ psq) {
    const int c = threadIdx.x;
    const int b = blockIdx.x;
    const float* p = M + (size_t)b * 128 * HID;
    float s = 0.f, q = 0.f;
    for (int r = 0; r < 128; ++r) {
        float v = p[(size_t)r * HID + c];
        s += v; q += v * v;
    }
    psum[b * HID + c] = s;
    psq [b * HID + c] = q;
}

__global__ __launch_bounds__(256) void bn_finalize_kernel(const float* __restrict__ psum,
                                                          const float* __restrict__ psq,
                                                          const float* __restrict__ gamma,
                                                          const float* __restrict__ beta,
                                                          float* __restrict__ scale,
                                                          float* __restrict__ shift) {
    const int c = threadIdx.x;
    float s = 0.f, q = 0.f;
    for (int b = 0; b < 256; ++b) { s += psum[b * HID + c]; q += psq[b * HID + c]; }
    const float mu   = s * (1.f / (float)BATCH);
    const float var  = q * (1.f / (float)BATCH) - mu * mu;
    const float rstd = rsqrtf(var + 1e-5f);
    const float sc   = gamma[c] * rstd;
    scale[c] = sc;
    shift[c] = beta[c] - mu * sc;
}

// ---- BN1+ReLU, log-map, 3 butterfly layers, exp-map ----------------------

__global__ __launch_bounds__(256) void butterfly_kernel(const float* __restrict__ H,
                                                        const float* __restrict__ sc1,
                                                        const float* __restrict__ sh1,
                                                        const float* __restrict__ P,
                                                        float* __restrict__ HT) {
    __shared__ float ssc[HID], ssh[HID];
    const int tid = threadIdx.x;
    ssc[tid] = sc1[tid];
    ssh[tid] = sh1[tid];
    __syncthreads();

    const int wave = tid >> 5, lane = tid & 31;
    const size_t row = (size_t)blockIdx.x * 8 + wave;
    const int e = lane * 8;
    const float* rp = H + row * HID + e;

    float u[8];
    {
        float4 f0 = *reinterpret_cast<const float4*>(rp);
        float4 f1 = *reinterpret_cast<const float4*>(rp + 4);
        u[0] = f0.x; u[1] = f0.y; u[2] = f0.z; u[3] = f0.w;
        u[4] = f1.x; u[5] = f1.y; u[6] = f1.z; u[7] = f1.w;
    }
#pragma unroll
    for (int i = 0; i < 8; ++i)
        u[i] = fmaxf(u[i] * ssc[e + i] + ssh[e + i], 0.f);

    const float sqrtc = 0.03162277660168379f;  // sqrt(0.001)

    // log_map0
    float ss = 0.f;
#pragma unroll
    for (int i = 0; i < 8; ++i) ss += u[i] * u[i];
#pragma unroll
    for (int m = 16; m > 0; m >>= 1) ss += __shfl_xor(ss, m, 32);
    {
        float sn = sqrtc * sqrtf(ss);
        sn = fminf(fmaxf(sn, 1e-7f), 1.0f - 1e-6f);
        const float lsc = atanhf(sn) / sn;
#pragma unroll
        for (int i = 0; i < 8; ++i) u[i] *= lsc;
    }

    // layer 0: bs=1, pairs (2i,2i+1); a=P[0:128], b=P[128:256]
#pragma unroll
    for (int i = 0; i < 8; i += 2) {
        const int blk = (e + i) >> 1;
        const float a = P[blk], b = P[128 + blk];
        const float x1 = u[i], x2 = u[i + 1];
        u[i]     =  a * x1 + b * x2;
        u[i + 1] = -b * x1 + a * x2;
    }
    // layer 1: bs=2, pairs distance 2; a=P[256:320], b=P[320:384]
#pragma unroll
    for (int g = 0; g < 2; ++g) {
        const int blk = (e + g * 4) >> 2;
        const float a = P[256 + blk], b = P[320 + blk];
#pragma unroll
        for (int i = 0; i < 2; ++i) {
            const int i1 = g * 4 + i, i2 = i1 + 2;
            const float x1 = u[i1], x2 = u[i2];
            u[i1] =  a * x1 + b * x2;
            u[i2] = -b * x1 + a * x2;
        }
    }
    // layer 2: bs=4, pairs distance 4; a=P[384:416], b=P[416:448]
    {
        const int blk = e >> 3;
        const float a = P[384 + blk], b = P[416 + blk];
#pragma unroll
        for (int i = 0; i < 4; ++i) {
            const float x1 = u[i], x2 = u[i + 4];
            u[i]     =  a * x1 + b * x2;
            u[i + 4] = -b * x1 + a * x2;
        }
    }

    // exp_map0
    ss = 0.f;
#pragma unroll
    for (int i = 0; i < 8; ++i) ss += u[i] * u[i];
#pragma unroll
    for (int m = 16; m > 0; m >>= 1) ss += __shfl_xor(ss, m, 32);
    {
        const float sn  = fmaxf(sqrtc * sqrtf(ss), 1e-7f);
        const float esc = tanhf(sn) / sn;
#pragma unroll
        for (int i = 0; i < 8; ++i) u[i] *= esc;
    }

    float* op = HT + row * HID + e;
    float4 o0 = {u[0], u[1], u[2], u[3]};
    float4 o1 = {u[4], u[5], u[6], u[7]};
    *reinterpret_cast<float4*>(op)     = o0;
    *reinterpret_cast<float4*>(op + 4) = o1;
}

// ---- GEMM2: OUT = relu(bn2(HT)) @ W2^T + b2 ------------------------------

__global__ __launch_bounds__(256) void gemm2_kernel(const float* __restrict__ HT,
                                                    const float* __restrict__ W2,
                                                    const float* __restrict__ B2,
                                                    const float* __restrict__ sc2,
                                                    const float* __restrict__ sh2,
                                                    float* __restrict__ OUT) {
    __shared__ __align__(16) __bf16 As[2][BM * LDT];
    __shared__ __align__(16) __bf16 Bs[2][BN * LDT];
    __shared__ float ssc[HID], ssh[HID];

    const int tid = threadIdx.x;
    ssc[tid] = sc2[tid];
    ssh[tid] = sh2[tid];

    const int m0   = blockIdx.x * BM;
    const int n0   = blockIdx.y * BN;
    const int wave = tid >> 5, lane = tid & 31;
    const int wm   = wave >> 1, wn = wave & 1;
    const int lm   = lane & 15, hs = lane >> 4;

    const int arow = tid >> 1, acb = (tid & 1) * 16;
    const int sofs = arow * LDT + acb;
    const bool browok = (n0 + arow) < OUT_DIM;

    float8 acc[2][4] = {};
    __syncthreads();   // ssc/ssh visible before first stageA

    // A staging with fused BN2+ReLU+cvt
    auto stageA = [&](int gk, bhalf8& r0, bhalf8& r1) {
        const float* p = HT + (size_t)(m0 + arow) * HID + gk;
        float v[16];
#pragma unroll
        for (int q = 0; q < 4; ++q) {
            float4 f = reinterpret_cast<const float4*>(p)[q];
            v[q * 4 + 0] = f.x; v[q * 4 + 1] = f.y;
            v[q * 4 + 2] = f.z; v[q * 4 + 3] = f.w;
        }
#pragma unroll
        for (int i = 0; i < 8; ++i) {
            r0[i] = f2bf(fmaxf(v[i]     * ssc[gk + i]     + ssh[gk + i],     0.f));
            r1[i] = f2bf(fmaxf(v[i + 8] * ssc[gk + i + 8] + ssh[gk + i + 8], 0.f));
        }
    };

    bhalf8 ra0, ra1, rb0, rb1;
    stageA(acb, ra0, ra1);
    stage16(W2, n0 + arow, acb, HID, HID, browok, rb0, rb1);
    *reinterpret_cast<bhalf8*>(&As[0][sofs])     = ra0;
    *reinterpret_cast<bhalf8*>(&As[0][sofs + 8]) = ra1;
    *reinterpret_cast<bhalf8*>(&Bs[0][sofs])     = rb0;
    *reinterpret_cast<bhalf8*>(&Bs[0][sofs + 8]) = rb1;
    __syncthreads();

    int cur = 0;
    for (int k0 = 0; k0 < HID; k0 += BK, cur ^= 1) {
        const int kn = k0 + BK;
        const bool has_next = kn < HID;

        if (has_next) {
            stageA(kn + acb, ra0, ra1);
            stage16(W2, n0 + arow, kn + acb, HID, HID, browok, rb0, rb1);
        }
        const int kp = k0 + 2 * BK;
        if (kp < HID) {
            __builtin_prefetch(HT + (size_t)(m0 + arow) * HID + kp + acb, 0, 1);
            if (browok)
                __builtin_prefetch(W2 + (size_t)(n0 + arow) * HID + kp + acb, 0, 1);
        }

        bhalf16 af[2], bf_[4];
#pragma unroll
        for (int i = 0; i < 2; ++i)
            af[i] = load_a_frag(&As[cur][(wm * 32 + i * 16 + lm) * LDT], hs);
#pragma unroll
        for (int j = 0; j < 4; ++j)
            bf_[j] = load_b_frag(&Bs[cur][(wn * 64 + j * 16 + lm) * LDT], hs);

#pragma unroll
        for (int i = 0; i < 2; ++i)
#pragma unroll
            for (int j = 0; j < 4; ++j)
                acc[i][j] = __builtin_amdgcn_wmma_f32_16x16x32_bf16(
                    false, af[i], false, bf_[j], (short)0, acc[i][j], false, false);

        if (has_next) {
            const int nxt = cur ^ 1;
            *reinterpret_cast<bhalf8*>(&As[nxt][sofs])     = ra0;
            *reinterpret_cast<bhalf8*>(&As[nxt][sofs + 8]) = ra1;
            *reinterpret_cast<bhalf8*>(&Bs[nxt][sofs])     = rb0;
            *reinterpret_cast<bhalf8*>(&Bs[nxt][sofs + 8]) = rb1;
            __syncthreads();
        }
    }

#pragma unroll
    for (int i = 0; i < 2; ++i)
#pragma unroll
        for (int j = 0; j < 4; ++j) {
            const int rb = m0 + wm * 32 + i * 16 + hs * 8;
            const int cb = n0 + wn * 64 + j * 16 + lm;
            if (cb < OUT_DIM) {
                const float bv = B2[cb];
#pragma unroll
                for (int r = 0; r < 8; ++r)
                    OUT[(size_t)(rb + r) * OUT_DIM + cb] = acc[i][j][r] + bv;
            }
        }
}

// ---- launcher ------------------------------------------------------------

extern "C" void kernel_launch(void* const* d_in, const int* in_sizes, int n_in,
                              void* d_out, int out_size, void* d_ws, size_t ws_size,
                              hipStream_t stream) {
    const float* x   = (const float*)d_in[0];
    const float* w1  = (const float*)d_in[1];
    const float* b1  = (const float*)d_in[2];
    const float* g1  = (const float*)d_in[3];
    const float* be1 = (const float*)d_in[4];
    const float* bfp = (const float*)d_in[5];
    const float* g2  = (const float*)d_in[6];
    const float* be2 = (const float*)d_in[7];
    const float* w2  = (const float*)d_in[8];
    const float* b2  = (const float*)d_in[9];
    float* out = (float*)d_out;

    float* ws  = (float*)d_ws;
    float* H   = ws;                                  // 32768*256
    float* HT  = ws + (size_t)BATCH * HID;            // 32768*256
    float* ps  = HT + (size_t)BATCH * HID;            // 256*256
    float* pq  = ps + 256 * HID;                      // 256*256
    float* sc1 = pq + 256 * HID;
    float* sh1 = sc1 + HID;
    float* sc2 = sh1 + HID;
    float* sh2 = sc2 + HID;

    gemm1_kernel<<<dim3(BATCH / BM, HID / BN), 256, 0, stream>>>(x, w1, b1, H);
    colstats_kernel<<<BATCH / 128, 256, 0, stream>>>(H, ps, pq);
    bn_finalize_kernel<<<1, 256, 0, stream>>>(ps, pq, g1, be1, sc1, sh1);
    butterfly_kernel<<<BATCH / 8, 256, 0, stream>>>(H, sc1, sh1, bfp, HT);
    colstats_kernel<<<BATCH / 128, 256, 0, stream>>>(HT, ps, pq);
    bn_finalize_kernel<<<1, 256, 0, stream>>>(ps, pq, g2, be2, sc2, sh2);
    gemm2_kernel<<<dim3(BATCH / BM, (OUT_DIM + BN - 1) / BN), 256, 0, stream>>>(
        HT, w2, b2, sc2, sh2, out);
}
